// GATEdgeNet_50568944943203
// MI455X (gfx1250) — matrix operand order repaired and checked by reading, not compiled, not run on hardware
//
#include <hip/hip_runtime.h>
#include <math.h>

typedef __bf16 bf16;
typedef __attribute__((ext_vector_type(16))) __bf16        v16bf;
typedef __attribute__((ext_vector_type(8)))  float         v8f;
typedef __attribute__((ext_vector_type(4)))  unsigned int  u32x4;

union FragBF {
  v16bf v;
  u32x4 q[2];
};

// ---------------------------------------------------------------------------
// Constants for this problem
// ---------------------------------------------------------------------------
#define N_NODES 50000
#define N_EDGES 800000
#define E_TOT   (2 * N_EDGES + N_NODES)   // fwd + reverse + self loops
#define GNN_IN  64
#define GNN_HID 128
#define GNN_OUT 64
#define MLP_K   272                       // (64+64)*2 + 16
#define MLP_KP  288                       // padded to multiple of 32
#define MLP_HID 128
#define NEG_SLOPE 0.2f
#define EPB     64                        // edges per block in the MLP kernel

// ---------------------------------------------------------------------------
// Small helper kernels (conversion / fill)
// ---------------------------------------------------------------------------
__global__ void fill_f32_kernel(float* __restrict__ p, float v, int n) {
  int i = blockIdx.x * blockDim.x + threadIdx.x;
  if (i < n) p[i] = v;
}

// dst[r*dst_stride + dst_off + c] = (bf16)src[r*cols + c]
__global__ void convert_rows_kernel(const float* __restrict__ src, bf16* __restrict__ dst,
                                    int rows, int cols, int dst_stride, int dst_off) {
  int i = blockIdx.x * blockDim.x + threadIdx.x;
  if (i >= rows * cols) return;
  int r = i / cols, c = i - r * cols;
  dst[(size_t)r * dst_stride + dst_off + c] = (bf16)src[(size_t)r * cols + c];
}

// src[R][C] -> dst[C][Kpad], dst[c][k] = (k < R) ? src[k][c] : 0
__global__ void transpose_convert_kernel(const float* __restrict__ src, bf16* __restrict__ dst,
                                         int R, int C, int Kpad) {
  int i = blockIdx.x * blockDim.x + threadIdx.x;
  if (i >= C * Kpad) return;
  int c = i / Kpad, k = i - c * Kpad;
  dst[i] = (k < R) ? (bf16)src[(size_t)k * C + c] : (bf16)0.f;
}

// ---------------------------------------------------------------------------
// bf16 WMMA GEMM:  C[M x N] = A[M x K] * B[K x N],  B given transposed Bt[N][K]
// block = 32 * (N/16) threads, grid.x = M/16 (M multiple of 16)
// ---------------------------------------------------------------------------
template <int K, int N>
__global__ void gemm_bf16_kernel(const bf16* __restrict__ A, const bf16* __restrict__ Bt,
                                 float* __restrict__ C) {
  const int lane = threadIdx.x & 31;
  const int wave = threadIdx.x >> 5;       // N tile index
  const int tile = blockIdx.x;             // M tile index
  const int rowA = tile * 16 + (lane & 15);
  const int colN = wave * 16 + (lane & 15);
  const int ka   = (lane >> 4) * 8;        // A: two 8-elem chunks (ISA 16-bit A layout)
  const int kb   = (lane >> 4) * 8;        // B: two 8-elem chunks (same k order as A)

  v8f acc = {0.f, 0.f, 0.f, 0.f, 0.f, 0.f, 0.f, 0.f};
#pragma unroll
  for (int k = 0; k < K; k += 32) {
    FragBF a, b;
    a.q[0] = *(const u32x4*)(A + (size_t)rowA * K + k + ka);
    a.q[1] = *(const u32x4*)(A + (size_t)rowA * K + k + ka + 16);
    b.q[0] = *(const u32x4*)(Bt + (size_t)colN * K + k + kb);
    b.q[1] = *(const u32x4*)(Bt + (size_t)colN * K + k + kb + 16);
    acc = __builtin_amdgcn_wmma_f32_16x16x32_bf16(false, a.v, false, b.v,
                                                  (short)0, acc, false, false);
  }
  const int rowC = tile * 16 + 8 * (lane >> 4);
#pragma unroll
  for (int i = 0; i < 8; ++i)
    C[(size_t)(rowC + i) * N + colN] = acc[i];
}

// ---------------------------------------------------------------------------
// Per-node attention logits: as[i] = h[i,:]·a_src, ad[i] = h[i,:]·a_dst
// ---------------------------------------------------------------------------
__global__ void alphas_kernel(const float* __restrict__ h, const float* __restrict__ a_src,
                              const float* __restrict__ a_dst, float* __restrict__ as,
                              float* __restrict__ ad, int n, int F) {
  int i = blockIdx.x * blockDim.x + threadIdx.x;
  if (i >= n) return;
  float ss = 0.f, sd = 0.f;
  for (int f = 0; f < F; ++f) {
    float v = h[(size_t)i * F + f];
    ss += v * a_src[f];
    sd += v * a_dst[f];
  }
  as[i] = ss;
  ad[i] = sd;
}

// ---------------------------------------------------------------------------
// Edge enumeration: e<E fwd, e<2E reverse, else self-loop
// ---------------------------------------------------------------------------
__device__ __forceinline__ void edge_src_dst(int e, const int* __restrict__ eidx,
                                             int* s, int* d) {
  if (e < N_EDGES) {
    *s = eidx[2 * e];
    *d = eidx[2 * e + 1];
  } else if (e < 2 * N_EDGES) {
    int e2 = e - N_EDGES;
    *s = eidx[2 * e2 + 1];
    *d = eidx[2 * e2];
  } else {
    *s = *d = e - 2 * N_EDGES;
  }
}

__device__ __forceinline__ void atomicMaxFloat(float* addr, float val) {
  if (val >= 0.f)
    atomicMax((int*)addr, __float_as_int(val));
  else
    atomicMin((unsigned int*)addr, (unsigned int)__float_as_int(val));
}

// e_val = leaky_relu(as[src]+ad[dst]);  emax[dst] = max(...)
__global__ void att_logits_kernel(const int* __restrict__ eidx, const float* __restrict__ as,
                                  const float* __restrict__ ad, float* __restrict__ e_val,
                                  float* __restrict__ emax) {
  int e = blockIdx.x * blockDim.x + threadIdx.x;
  if (e >= E_TOT) return;
  int s, d;
  edge_src_dst(e, eidx, &s, &d);
  float v = as[s] + ad[d];
  v = (v > 0.f) ? v : NEG_SLOPE * v;
  e_val[e] = v;
  atomicMaxFloat(&emax[d], v);
}

// w = exp(e_val - emax[dst]); denom[dst] += w; e_val <- w (in place)
__global__ void att_softmax_kernel(const int* __restrict__ eidx, float* __restrict__ e_val,
                                   const float* __restrict__ emax, float* __restrict__ denom) {
  int e = blockIdx.x * blockDim.x + threadIdx.x;
  if (e >= E_TOT) return;
  int s, d;
  edge_src_dst(e, eidx, &s, &d);
  float w = expf(e_val[e] - emax[d]);
  e_val[e] = w;
  atomicAdd(&denom[d], w);
}

// agg[dst,:] += (w/denom[dst]) * h[src,:]   (block per edge, thread per feature)
__global__ void att_aggregate_kernel(const int* __restrict__ eidx, const float* __restrict__ w,
                                     const float* __restrict__ denom, const float* __restrict__ h,
                                     float* __restrict__ agg, int F) {
  int e = blockIdx.x;
  int f = threadIdx.x;
  int s, d;
  edge_src_dst(e, eidx, &s, &d);
  float alpha = w[e] / (denom[d] + 1e-16f);
  atomicAdd(&agg[(size_t)d * F + f], alpha * h[(size_t)s * F + f]);
}

// out_bf16[r*stride + off + c] = relu(agg[r*F+c] + bias[c])
__global__ void finalize_kernel(const float* __restrict__ agg, const float* __restrict__ bias,
                                bf16* __restrict__ dst, int n, int F, int dst_stride,
                                int dst_off) {
  int i = blockIdx.x * blockDim.x + threadIdx.x;
  if (i >= n * F) return;
  int r = i / F, c = i - r * F;
  float v = agg[i] + bias[c];
  v = (v > 0.f) ? v : 0.f;
  dst[(size_t)r * dst_stride + dst_off + c] = (bf16)v;
}

// ---------------------------------------------------------------------------
// Fused edge MLP: gather [xf[row] | xf[col] | dist[e] | 0-pad] (K=288) and run
// 272->128 (WMMA bf16) -> ReLU -> 128->1 -> sigmoid.
// 64 edges per block (4 M-tiles), 8 waves (one per 16-wide hidden tile).
// Each wave keeps its B (weight) fragment in registers across the 4 M-tiles,
// cutting load traffic per WMMA from 64 B/lane to 40 B/lane.
// ---------------------------------------------------------------------------
__device__ __forceinline__ u32x4 load_edge_chunk(int kk, int row, int col, int e,
                                                 const bf16* __restrict__ xf,
                                                 const bf16* __restrict__ dist) {
  if (kk < 128) return *(const u32x4*)(xf + (size_t)row * 128 + kk);
  if (kk < 256) return *(const u32x4*)(xf + (size_t)col * 128 + (kk - 128));
  if (kk < 272) return *(const u32x4*)(dist + (size_t)e * 16 + (kk - 256));
  u32x4 z = {0u, 0u, 0u, 0u};
  return z;
}

__global__ void edge_mlp_kernel(const int* __restrict__ eidx, const bf16* __restrict__ xf,
                                const bf16* __restrict__ dist, const bf16* __restrict__ W1t,
                                const float* __restrict__ b1, const float* __restrict__ W2,
                                const float* __restrict__ b2, float* __restrict__ out) {
  __shared__ float lds_h[EPB][MLP_HID];   // 32 KB
  __shared__ float lds_p[EPB][4];

  const int lane = threadIdx.x & 31;
  const int wave = threadIdx.x >> 5;   // hidden tile 0..7
  const int tile = blockIdx.x;         // 64 edges
  const int m    = lane & 15;
  const int hi   = lane >> 4;
  const int ka   = hi * 8;
  const int kb   = hi * 8;
  const int colN = wave * 16 + m;

  int ev[4], row[4], col[4];
#pragma unroll
  for (int t = 0; t < 4; ++t) {
    ev[t]  = tile * EPB + t * 16 + m;
    row[t] = eidx[2 * ev[t]];
    col[t] = eidx[2 * ev[t] + 1];
  }

  v8f acc[4];
#pragma unroll
  for (int t = 0; t < 4; ++t) acc[t] = (v8f){0.f, 0.f, 0.f, 0.f, 0.f, 0.f, 0.f, 0.f};

#pragma unroll
  for (int k = 0; k < MLP_KP; k += 32) {
    FragBF b;                     // weight fragment: loaded once, reused 4x
    b.q[0] = *(const u32x4*)(W1t + (size_t)colN * MLP_KP + k + kb);
    b.q[1] = *(const u32x4*)(W1t + (size_t)colN * MLP_KP + k + kb + 16);
#pragma unroll
    for (int t = 0; t < 4; ++t) {
      FragBF a;
      a.q[0] = load_edge_chunk(k + ka, row[t], col[t], ev[t], xf, dist);
      a.q[1] = load_edge_chunk(k + ka + 16, row[t], col[t], ev[t], xf, dist);
      acc[t] = __builtin_amdgcn_wmma_f32_16x16x32_bf16(false, a.v, false, b.v,
                                                       (short)0, acc[t], false, false);
    }
  }

  // bias + ReLU, stash hidden tiles to LDS.  C layout: VGPR i holds row i + 8*hi.
  const float bias1 = b1[colN];
#pragma unroll
  for (int t = 0; t < 4; ++t) {
    const int rowH = t * 16 + 8 * hi;
#pragma unroll
    for (int i = 0; i < 8; ++i) {
      float h = acc[t][i] + bias1;
      lds_h[rowH + i][colN] = (h > 0.f) ? h : 0.f;
    }
  }
  __syncthreads();

  // layer 2: 128 -> 1 for 64 edges; 4 threads per edge, 32 columns each
  const int em = threadIdx.x >> 2;   // edge row 0..63
  const int q  = threadIdx.x & 3;    // quarter 0..3
  float p = 0.f;
#pragma unroll
  for (int j = 0; j < 32; ++j) {
    int n = q * 32 + j;
    p += lds_h[em][n] * W2[n];
  }
  lds_p[em][q] = p;
  __syncthreads();

  if (threadIdx.x < EPB) {
    float s = b2[0] + lds_p[threadIdx.x][0] + lds_p[threadIdx.x][1] +
              lds_p[threadIdx.x][2] + lds_p[threadIdx.x][3];
    out[tile * EPB + threadIdx.x] = 1.f / (1.f + expf(-s));
  }
}

// ---------------------------------------------------------------------------
// Launcher
// ---------------------------------------------------------------------------
static inline int cdiv(int a, int b) { return (a + b - 1) / b; }

extern "C" void kernel_launch(void* const* d_in, const int* in_sizes, int n_in,
                              void* d_out, int out_size, void* d_ws, size_t ws_size,
                              hipStream_t stream) {
  const float* x        = (const float*)d_in[0];
  const int*   eidx     = (const int*)d_in[1];
  const float* edist    = (const float*)d_in[2];
  const float* W_g1     = (const float*)d_in[3];
  const float* a1_src   = (const float*)d_in[4];
  const float* a1_dst   = (const float*)d_in[5];
  const float* b_g1     = (const float*)d_in[6];
  const float* W_g2     = (const float*)d_in[7];
  const float* a2_src   = (const float*)d_in[8];
  const float* a2_dst   = (const float*)d_in[9];
  const float* b_g2     = (const float*)d_in[10];
  const float* W_m1     = (const float*)d_in[11];
  const float* b_m1     = (const float*)d_in[12];
  const float* W_m2     = (const float*)d_in[13];
  const float* b_m2     = (const float*)d_in[14];
  float*       out      = (float*)d_out;

  // ---- bump allocator over d_ws (256B aligned slices) ----
  char* ws = (char*)d_ws;
  size_t off = 0;
  auto alloc = [&](size_t bytes) -> void* {
    void* p = ws + off;
    off += (bytes + 255) & ~(size_t)255;
    return p;
  };

  bf16*  x_bf    = (bf16*)alloc((size_t)N_NODES * GNN_IN * 2);
  bf16*  xf_bf   = (bf16*)alloc((size_t)N_NODES * 128 * 2);
  bf16*  Wg1t    = (bf16*)alloc((size_t)GNN_HID * GNN_IN * 2);
  bf16*  Wg2t    = (bf16*)alloc((size_t)GNN_OUT * GNN_HID * 2);
  bf16*  Wm1t    = (bf16*)alloc((size_t)MLP_HID * MLP_KP * 2);
  bf16*  dist_bf = (bf16*)alloc((size_t)N_EDGES * 16 * 2);
  float* h1      = (float*)alloc((size_t)N_NODES * GNN_HID * 4);
  bf16*  f1_bf   = (bf16*)alloc((size_t)N_NODES * GNN_HID * 2);
  float* h2      = (float*)alloc((size_t)N_NODES * GNN_OUT * 4);
  float* as      = (float*)alloc((size_t)N_NODES * 4);
  float* ad      = (float*)alloc((size_t)N_NODES * 4);
  float* e_val   = (float*)alloc((size_t)E_TOT * 4);
  float* emax    = (float*)alloc((size_t)N_NODES * 4);
  float* denom   = (float*)alloc((size_t)N_NODES * 4);
  float* agg1    = (float*)alloc((size_t)N_NODES * GNN_HID * 4);
  float* agg2    = (float*)alloc((size_t)N_NODES * GNN_OUT * 4);

  const int B = 256;

  // ---- staging: bf16 conversions + weight transposes ----
  convert_rows_kernel<<<cdiv(N_NODES * GNN_IN, B), B, 0, stream>>>(x, x_bf, N_NODES, GNN_IN, GNN_IN, 0);
  convert_rows_kernel<<<cdiv(N_NODES * GNN_IN, B), B, 0, stream>>>(x, xf_bf, N_NODES, GNN_IN, 128, 0);
  convert_rows_kernel<<<cdiv(N_EDGES * 16, B), B, 0, stream>>>(edist, dist_bf, N_EDGES, 16, 16, 0);
  transpose_convert_kernel<<<cdiv(GNN_HID * GNN_IN, B), B, 0, stream>>>(W_g1, Wg1t, GNN_IN, GNN_HID, GNN_IN);
  transpose_convert_kernel<<<cdiv(GNN_OUT * GNN_HID, B), B, 0, stream>>>(W_g2, Wg2t, GNN_HID, GNN_OUT, GNN_HID);
  transpose_convert_kernel<<<cdiv(MLP_HID * MLP_KP, B), B, 0, stream>>>(W_m1, Wm1t, MLP_K, MLP_HID, MLP_KP);

  // ================= GAT layer 1 (64 -> 128) =================
  gemm_bf16_kernel<GNN_IN, GNN_HID><<<N_NODES / 16, 32 * (GNN_HID / 16), 0, stream>>>(x_bf, Wg1t, h1);
  alphas_kernel<<<cdiv(N_NODES, B), B, 0, stream>>>(h1, a1_src, a1_dst, as, ad, N_NODES, GNN_HID);
  fill_f32_kernel<<<cdiv(N_NODES, B), B, 0, stream>>>(emax, -__builtin_inff(), N_NODES);
  fill_f32_kernel<<<cdiv(N_NODES, B), B, 0, stream>>>(denom, 0.f, N_NODES);
  fill_f32_kernel<<<cdiv(N_NODES * GNN_HID, B), B, 0, stream>>>(agg1, 0.f, N_NODES * GNN_HID);
  att_logits_kernel<<<cdiv(E_TOT, B), B, 0, stream>>>(eidx, as, ad, e_val, emax);
  att_softmax_kernel<<<cdiv(E_TOT, B), B, 0, stream>>>(eidx, e_val, emax, denom);
  att_aggregate_kernel<<<E_TOT, GNN_HID, 0, stream>>>(eidx, e_val, denom, h1, agg1, GNN_HID);
  finalize_kernel<<<cdiv(N_NODES * GNN_HID, B), B, 0, stream>>>(agg1, b_g1, f1_bf, N_NODES, GNN_HID, GNN_HID, 0);

  // ================= GAT layer 2 (128 -> 64) =================
  gemm_bf16_kernel<GNN_HID, GNN_OUT><<<N_NODES / 16, 32 * (GNN_OUT / 16), 0, stream>>>(f1_bf, Wg2t, h2);
  alphas_kernel<<<cdiv(N_NODES, B), B, 0, stream>>>(h2, a2_src, a2_dst, as, ad, N_NODES, GNN_OUT);
  fill_f32_kernel<<<cdiv(N_NODES, B), B, 0, stream>>>(emax, -__builtin_inff(), N_NODES);
  fill_f32_kernel<<<cdiv(N_NODES, B), B, 0, stream>>>(denom, 0.f, N_NODES);
  fill_f32_kernel<<<cdiv(N_NODES * GNN_OUT, B), B, 0, stream>>>(agg2, 0.f, N_NODES * GNN_OUT);
  att_logits_kernel<<<cdiv(E_TOT, B), B, 0, stream>>>(eidx, as, ad, e_val, emax);
  att_softmax_kernel<<<cdiv(E_TOT, B), B, 0, stream>>>(eidx, e_val, emax, denom);
  att_aggregate_kernel<<<E_TOT, GNN_OUT, 0, stream>>>(eidx, e_val, denom, h2, agg2, GNN_OUT);
  // f2 goes into the high 64 columns of xf (low 64 = x, already converted)
  finalize_kernel<<<cdiv(N_NODES * GNN_OUT, B), B, 0, stream>>>(agg2, b_g2, xf_bf, N_NODES, GNN_OUT, 128, GNN_IN);

  // ================= fused edge MLP (dominant compute) =================
  edge_mlp_kernel<<<N_EDGES / EPB, 256, 0, stream>>>(eidx, xf_bf, dist_bf, Wm1t, b_m1, W_m2, b_m2, out);

  (void)in_sizes; (void)n_in; (void)out_size; (void)ws_size;
}
